// NGCNNetwork_81810537054874
// MI455X (gfx1250) — matrix operand order repaired and checked by reading, not compiled, not run on hardware
//
#include <hip/hip_runtime.h>
#include <hip/hip_bf16.h>
#include <math.h>

typedef __bf16 bf16_t;
typedef __attribute__((ext_vector_type(16))) __bf16 v16bf;
typedef __attribute__((ext_vector_type(8)))  __bf16 v8bf;
typedef __attribute__((ext_vector_type(8)))  float  v8f;

// ---------------------------------------------------------------------------
// Kernel 1: convert+swizzle W = [w1|w2|w3] (f32 [F,64] each) into bf16 B-fragment
// layout for v_wmma_f32_16x16x32_bf16:
//   fragment (kt,nt) is 32 lanes x 16 bf16 (32B contiguous per lane)
//   lane = nn + 16*((kk>>3)&1), element j = (kk&7) + 8*(kk>=16) holds B[k=kt*32+kk][n=nt*16+nn]
// Also builds concatenated bias vector bcat[192].
// ---------------------------------------------------------------------------
__global__ void prep_w_kernel(const float* __restrict__ w1, const float* __restrict__ b1,
                              const float* __restrict__ w2, const float* __restrict__ b2,
                              const float* __restrict__ w3, const float* __restrict__ b3,
                              bf16_t* __restrict__ Wcs, float* __restrict__ bcat, int F) {
  int t = blockIdx.x * blockDim.x + threadIdx.x;
  int total = F * 192;
  if (t >= total) return;
  int k = t / 192;
  int n = t - k * 192;
  int br = n >> 6;            // branch 0..2
  int within = n & 63;        // column inside branch
  const float* w = (br == 0) ? w1 : (br == 1) ? w2 : w3;
  float val = w[(size_t)k * 64 + within];
  int kt = k >> 5, kk = k & 31;
  int nt = n >> 4, nn = n & 15;
  int lane = nn + (((kk >> 3) & 1) << 4);
  int j = (kk & 7) + ((kk >> 4) << 3);
  size_t dst = ((size_t)(kt * 12 + nt) * 32 + lane) * 16 + j;
  Wcs[dst] = (bf16_t)val;
  if (t < 192) {  // k==0, n==t
    const float* b = (br == 0) ? b1 : (br == 1) ? b2 : b3;
    bcat[n] = b[within];
  }
}

// ---------------------------------------------------------------------------
// Kernel 2: densify sparse features into bf16 Xd[N,F] (zeroed beforehand).
// global_atomic_pk_add_bf16 makes duplicate (row,col) entries sum like
// jax segment_sum.
// ---------------------------------------------------------------------------
__global__ void scatter_feat_kernel(const int* __restrict__ fi, const float* __restrict__ fv,
                                    bf16_t* __restrict__ Xd, int nnz, int F) {
  int t = blockIdx.x * blockDim.x + threadIdx.x;
  if (t >= nnz) return;
  int r = fi[t];
  int c = fi[nnz + t];
  float v = fv[t];
  size_t idx = (size_t)r * F + c;
  bf16_t bv = (bf16_t)v;
  unsigned int bits = (unsigned int)__builtin_bit_cast(unsigned short, bv);
  unsigned int data = bits << ((unsigned)(idx & 1) * 16);   // other half adds +0.0
  unsigned long long addr = (unsigned long long)(Xd + (idx & ~(size_t)1));
  asm volatile("global_atomic_pk_add_bf16 %0, %1, off" :: "v"(addr), "v"(data) : "memory");
}

// ---------------------------------------------------------------------------
// Kernel 3: H[N,192] = relu(Xd[N,F](bf16) @ Wcs(bf16 swizzled) + bcat)
// 4 waves/block, one 16-row m-tile per wave, all 12 n-tiles per wave.
// B staged via async-to-LDS (double buffered, ASYNCcnt). launch_bounds(128,1)
// unlocks the full VGPR file so all 12 B fragments live in distinct registers:
// the 24 ds_load_b128 issue as one clause and WMMAs overlap DS latency.
// A ping-pongs between two named registers via a 2x-unrolled k-loop.
// ---------------------------------------------------------------------------
#define B_KSTEP_BYTES 12288           // 12 frags * 32 lanes * 32B
#define B_KSTEP_ELEMS 6144

__global__ __launch_bounds__(128, 1) void gemm_wmma_kernel(const bf16_t* __restrict__ Xd,
                                                           const bf16_t* __restrict__ Wcs,
                                                           const float* __restrict__ bcat,
                                                           float* __restrict__ H,
                                                           int N, int F) {
  __shared__ __align__(32) bf16_t smemB[2 * B_KSTEP_ELEMS];   // 24 KB double buffer

  const int tid  = threadIdx.x;
  const int lane = tid & 31;
  const int wv   = tid >> 5;
  const int ntilesM = (N + 15) >> 4;
  const int rt = blockIdx.x * 4 + wv;
  const bool active = (rt < ntilesM);           // wave-uniform
  const int m0 = (active ? rt : 0) << 4;
  const int KT = F >> 5;

  const unsigned ldsBase = (unsigned)(unsigned long long)(uintptr_t)smemB;
  const unsigned long long wbase = (unsigned long long)(uintptr_t)Wcs;

  // Issue one k-step's worth of B into LDS buffer `buf` (6 async b128 per wave).
  auto issueB = [&](int buf, int kt) {
#pragma unroll
    for (int c = 0; c < 6; ++c) {
      unsigned lds = ldsBase + (unsigned)buf * B_KSTEP_BYTES + c * 2048 + tid * 16;
      unsigned long long ga = wbase + (unsigned long long)kt * B_KSTEP_BYTES + c * 2048 + tid * 16;
      asm volatile("global_load_async_to_lds_b128 %0, %1, off" :: "v"(lds), "v"(ga) : "memory");
    }
  };

  int row = m0 + (lane & 15);
  if (row >= N) row = N - 1;                    // safe clamp (N%16==0 normally)
  const bf16_t* aptr = Xd + (size_t)row * F + ((lane >> 4) << 3);

  auto loadA = [&](int kt) -> v16bf {
    const bf16_t* ap = aptr + (kt << 5);
    v8bf alo = *(const v8bf*)(ap);              // K {0..7}   ({8..15} for hi lanes)
    v8bf ahi = *(const v8bf*)(ap + 16);         // K {16..23} ({24..31})
    return __builtin_shufflevector(alo, ahi,
        0, 1, 2, 3, 4, 5, 6, 7, 8, 9, 10, 11, 12, 13, 14, 15);
  };

  v8f acc[12];
  v8f zero = {0.f, 0.f, 0.f, 0.f, 0.f, 0.f, 0.f, 0.f};
#pragma unroll
  for (int i = 0; i < 12; ++i) acc[i] = zero;

  // One pipeline stage: consume buffer (kt&1); fill a_fill with A(kt+1);
  // refill the just-consumed LDS buffer with B(kt+2).
  auto step = [&](int kt, v16bf& a_use, v16bf& a_fill) {
    // 6 async loads per group, in-order completion: <=6 outstanding means the
    // group for this buffer has landed (0 on the last k-step: no newer group).
    if (kt + 1 < KT) {
      asm volatile("s_wait_asynccnt 0x6" ::: "memory");
    } else {
      asm volatile("s_wait_asynccnt 0x0" ::: "memory");
    }
    __syncthreads();                            // publish all waves' LDS writes

    if (active) {
      const bf16_t* bufp = smemB + (size_t)(kt & 1) * B_KSTEP_ELEMS + (size_t)lane * 16;
      v16bf b[12];
#pragma unroll
      for (int nt = 0; nt < 12; ++nt)           // one clause of 24 ds_load_b128
        b[nt] = *(const v16bf*)(bufp + nt * 512);
      if (kt + 1 < KT) {
        a_fill = loadA(kt + 1);
        __builtin_prefetch(aptr + ((kt + 2) << 5), 0, 1);
      }
#pragma unroll
      for (int nt = 0; nt < 12; ++nt)
        acc[nt] = __builtin_amdgcn_wmma_f32_16x16x32_bf16(
            false, a_use, false, b[nt], (short)0, acc[nt], false, false);
    }

    __syncthreads();                            // all waves done reading buf
    if (kt + 2 < KT) issueB(kt & 1, kt + 2);    // refill the buffer just consumed
  };

  // Prologue: fill both buffers, preload A(0).
  issueB(0, 0);
  if (KT > 1) issueB(1, 1);
  v16bf aP, aQ;
  if (active) aP = loadA(0);

  int kt = 0;
  for (; kt + 1 < KT; kt += 2) {                // KT is even for F=2048
    step(kt,     aP, aQ);
    step(kt + 1, aQ, aP);
  }
  if (kt < KT) step(kt, aP, aQ);                // odd-KT tail

  if (!active) return;

  // Epilogue: C/D layout: lane(0-15)->M=r,N=lane ; lane(16-31)->M=8+r,N=lane-16
  const int n  = lane & 15;
  const int mh = (lane >> 4) << 3;
#pragma unroll
  for (int nt = 0; nt < 12; ++nt) {
    const int c = nt * 16 + n;
    const float bias = bcat[c];
#pragma unroll
    for (int r = 0; r < 8; ++r) {
      int m = m0 + mh + r;
      if (m < N) {
        float v = acc[nt][r] + bias;
        H[(size_t)m * 192 + c] = v > 0.f ? v : 0.f;
      }
    }
  }
}

// ---------------------------------------------------------------------------
// Kernel 4: one adjacency hop: xout[dst,0:64] += val * xin[src,0:64]
// (segment_sum over rows). xout must be pre-zeroed. 64 threads per edge.
// ---------------------------------------------------------------------------
__global__ void spmm_hop_kernel(const int* __restrict__ rows, const int* __restrict__ cols,
                                const float* __restrict__ vals, const float* __restrict__ xin,
                                int instride, float* __restrict__ xout, int E) {
  long long t = (long long)blockIdx.x * blockDim.x + threadIdx.x;
  long long e = t >> 6;
  int c = (int)(t & 63);
  if (e >= E) return;
  int dst = rows[e];
  int src = cols[e];
  float v = vals[e];
  float contrib = v * xin[(size_t)src * instride + c];
  __hip_atomic_fetch_add(&xout[(size_t)dst * 64 + c], contrib,
                         __ATOMIC_RELAXED, __HIP_MEMORY_SCOPE_AGENT);
}

// ---------------------------------------------------------------------------
// Kernel 5: logits = [H[:,0:64] | y2 | y3] @ fcw + fcb ; log_softmax rows.
// One wave32 per node; lane covers classes {lane, lane+32 (lanes 0-7)}.
// ---------------------------------------------------------------------------
__global__ __launch_bounds__(128) void fc_logsoftmax_kernel(
    const float* __restrict__ H, const float* __restrict__ y2, const float* __restrict__ y3,
    const float* __restrict__ fcw, const float* __restrict__ fcb,
    float* __restrict__ out, int N) {
  int node = blockIdx.x * 4 + (threadIdx.x >> 5);
  int lane = threadIdx.x & 31;
  if (node >= N) return;
  const float* h0 = H  + (size_t)node * 192;   // branch 1 (cols 0..63)
  const float* h1 = y2 + (size_t)node * 64;    // branch 2 after 1 hop
  const float* h2 = y3 + (size_t)node * 64;    // branch 3 after 2 hops
  const bool hi = (lane < 8);
  float a0 = fcb[lane];
  float a1 = hi ? fcb[lane + 32] : 0.f;
  for (int k = 0; k < 64; ++k) {
    float hv = h0[k];
    a0 += hv * fcw[k * 40 + lane];
    if (hi) a1 += hv * fcw[k * 40 + lane + 32];
  }
  for (int k = 0; k < 64; ++k) {
    float hv = h1[k];
    a0 += hv * fcw[(64 + k) * 40 + lane];
    if (hi) a1 += hv * fcw[(64 + k) * 40 + lane + 32];
  }
  for (int k = 0; k < 64; ++k) {
    float hv = h2[k];
    a0 += hv * fcw[(128 + k) * 40 + lane];
    if (hi) a1 += hv * fcw[(128 + k) * 40 + lane + 32];
  }
  float m = a0;
  if (hi) m = fmaxf(m, a1);
  for (int off = 16; off; off >>= 1) m = fmaxf(m, __shfl_xor(m, off, 32));
  float s = expf(a0 - m) + (hi ? expf(a1 - m) : 0.f);
  for (int off = 16; off; off >>= 1) s += __shfl_xor(s, off, 32);
  float lz = m + logf(s);
  out[(size_t)node * 40 + lane] = a0 - lz;
  if (hi) out[(size_t)node * 40 + lane + 32] = a1 - lz;
}

// ---------------------------------------------------------------------------
extern "C" void kernel_launch(void* const* d_in, const int* in_sizes, int n_in,
                              void* d_out, int out_size, void* d_ws, size_t ws_size,
                              hipStream_t stream) {
  const int*   adj_idx  = (const int*)d_in[0];
  const float* adj_val  = (const float*)d_in[1];
  const int*   feat_idx = (const int*)d_in[2];
  const float* feat_val = (const float*)d_in[3];
  const float* w1 = (const float*)d_in[4];
  const float* b1 = (const float*)d_in[5];
  const float* w2 = (const float*)d_in[6];
  const float* b2 = (const float*)d_in[7];
  const float* w3 = (const float*)d_in[8];
  const float* b3 = (const float*)d_in[9];
  const float* fcw = (const float*)d_in[10];
  const float* fcb = (const float*)d_in[11];

  const int E    = in_sizes[1];                 // 800000
  const int NNZ  = in_sizes[3];                 // 1600000
  const int HIDd = in_sizes[5];                 // 64
  const int F    = in_sizes[4] / HIDd;          // 2048
  const int C    = in_sizes[11];                // 40
  const int N    = out_size / C;                // 50000
  float* out = (float*)d_out;

  // Workspace carve-up
  char* ws = (char*)d_ws;
  size_t off = 0;
  auto carve = [&](size_t bytes) -> char* {
    char* p = ws + off;
    off += (bytes + 255) & ~(size_t)255;
    return p;
  };
  size_t xd_bytes = (size_t)N * F * sizeof(bf16_t);          // 204.8 MB
  bf16_t* Xd  = (bf16_t*)carve(xd_bytes);
  bf16_t* Wcs = (bf16_t*)carve((size_t)F * 192 * sizeof(bf16_t));
  float*  bcat = (float*)carve(192 * sizeof(float));
  float*  H   = (float*)carve((size_t)N * 192 * sizeof(float));
  size_t hopb = (size_t)N * 64 * sizeof(float);
  char*   hop0 = carve(3 * hopb);                             // y2|t3|y3 contiguous
  float*  y2 = (float*)hop0;
  float*  t3 = (float*)(hop0 + hopb);
  float*  y3 = (float*)(hop0 + 2 * hopb);

  // Zero accumulation targets (stream-ordered, graph-capturable)
  hipMemsetAsync(Xd, 0, xd_bytes, stream);
  hipMemsetAsync(hop0, 0, 3 * hopb, stream);

  // 1) weight swizzle + bias concat
  prep_w_kernel<<<(F * 192 + 255) / 256, 256, 0, stream>>>(
      w1, b1, w2, b2, w3, b3, Wcs, bcat, F);

  // 2) densify features to bf16 (atomic pk-add handles duplicate indices)
  scatter_feat_kernel<<<(NNZ + 255) / 256, 256, 0, stream>>>(
      feat_idx, feat_val, Xd, NNZ, F);

  // 3) fused 3-branch GEMM + bias + relu : H[N,192]
  int ntilesM = (N + 15) / 16;
  gemm_wmma_kernel<<<(ntilesM + 3) / 4, 128, 0, stream>>>(Xd, Wcs, bcat, H, N, F);

  // 4) adjacency hops (sequential by stream order)
  long long hop_threads = (long long)E * 64;
  int hop_blocks = (int)((hop_threads + 255) / 256);
  spmm_hop_kernel<<<hop_blocks, 256, 0, stream>>>(adj_idx, adj_idx + E, adj_val,
                                                  H + 64, 192, y2, E);   // branch2: 1 hop
  spmm_hop_kernel<<<hop_blocks, 256, 0, stream>>>(adj_idx, adj_idx + E, adj_val,
                                                  H + 128, 192, t3, E);  // branch3: hop 1
  spmm_hop_kernel<<<hop_blocks, 256, 0, stream>>>(adj_idx, adj_idx + E, adj_val,
                                                  t3, 64, y3, E);        // branch3: hop 2

  // 5) FC + log_softmax
  fc_logsoftmax_kernel<<<(N + 3) / 4, 128, 0, stream>>>(H, y2, y3, fcw, fcb, out, N);
}
